// FusedQKVLinear_678604833227
// MI455X (gfx1250) — matrix operand order repaired and checked by reading, not compiled, not run on hardware
//
#include <hip/hip_runtime.h>
#include <hip/hip_bf16.h>

// ---------------- CDNA5 WMMA types (probe-confirmed signatures) ----------------
typedef __attribute__((ext_vector_type(16))) __bf16 v16bf;
typedef __attribute__((ext_vector_type(8)))  __bf16 v8bf;
typedef __attribute__((ext_vector_type(8)))  float  v8f;

#define D_DIM   4096
#define QO      4096
#define KO      1024
#define VO      1024
#define MTILE   128      // M (token) tile per workgroup
#define NTILE   256      // N (output) tile per workgroup
#define KSTEP   32       // K per iteration (= WMMA bf16 K)
#define NKITERS (D_DIM / KSTEP)   // 128
#define ROWE    40       // LDS row stride in bf16 elems: 32 data + 8 pad = 80B (bank-conflict free)

__device__ __forceinline__ v8bf cvt8(const float* w) {
    v8bf r;
#pragma unroll
    for (int j = 0; j < 8; ++j) r[j] = (__bf16)w[j];
    return r;
}

__global__ __launch_bounds__(256)
void qkv_wmma_bf16_kernel(const float* __restrict__ x,
                          const int*   __restrict__ qc, const float* __restrict__ qs, const float* __restrict__ qcb,
                          const int*   __restrict__ kc, const float* __restrict__ ks, const float* __restrict__ kcb,
                          const int*   __restrict__ vc, const float* __restrict__ vs, const float* __restrict__ vcb,
                          float* __restrict__ out, int T)
{
    // LDS: double-buffered bf16 A (x, 128x32) and B (dequantized W, 256x32) + pair-LUT
    __shared__ __bf16 Ab[2][MTILE * ROWE];   // 2 x 10,240 B
    __shared__ __bf16 Bb[2][NTILE * ROWE];   // 2 x 20,480 B
    __shared__ float2 lut[256];              // (cb[i], cb[j]) pairs, 2 KB  -> total 63,488 B

    const int tid  = threadIdx.x;
    const int lane = tid & 31;
    const int wave = tid >> 5;
    const int wm   = wave & 1;      // 0..1 -> 64-row M band
    const int wn   = wave >> 1;     // 0..3 -> 64-col N band

    const int n0 = blockIdx.x * NTILE;  // global fused output column
    const int t0 = blockIdx.y * MTILE;  // token row

    // Segment resolution (4096 and 5120 are multiples of 256: tiles never straddle)
    const int* codes; const float* scales; const float* cb;
    float* outp; int segW; int nloc;
    if (n0 < QO)           { codes = qc; scales = qs; cb = qcb; outp = out;                         segW = QO; nloc = n0; }
    else if (n0 < QO + KO) { codes = kc; scales = ks; cb = kcb; outp = out + (size_t)T * QO;        segW = KO; nloc = n0 - QO; }
    else                   { codes = vc; scales = vs; cb = vcb; outp = out + (size_t)T * (QO + KO); segW = VO; nloc = n0 - QO - KO; }

    // Build 256-entry pair LUT: one ds_load_b64 dequantizes two codes at once
    lut[tid] = make_float2(cb[tid >> 4], cb[tid & 15]);
    __syncthreads();

    // Staging assignment:
    //   B: 1 thread per output row (256 rows), 32 codes each (full K step)
    //   A: 2 threads per token row (128 rows), 16 floats each
    const int brow = tid;                // 0..255
    const int ogB  = nloc + brow;
    const int*   crow = codes + (size_t)ogB * D_DIM;
    const float* srow = scales + (size_t)ogB * (D_DIM / 128);

    const int arow = tid >> 1;           // 0..127
    const int half = tid & 1;
    const float* xrow = x + (size_t)(t0 + arow) * D_DIM + half * 16;

    float4 xr[4]; int4 cr[8]; float scl;
    auto gload = [&](int kt) {
        const int*   cp = crow + kt * KSTEP;
        const float* xp = xrow + kt * KSTEP;
#pragma unroll
        for (int i = 0; i < 8; ++i) cr[i] = ((const int4*)cp)[i];
#pragma unroll
        for (int i = 0; i < 4; ++i) xr[i] = ((const float4*)xp)[i];
        scl = srow[(kt * KSTEP) >> 7];   // one quant group per aligned 32-k step
    };

    v8f acc[4][4] = {};

    gload(0);

    for (int kt = 0; kt < NKITERS; ++kt) {
        const int buf = kt & 1;

        // ---- dequant weights + convert x, store bf16 tiles to LDS ----
        {
            int c[32];
#pragma unroll
            for (int i = 0; i < 8; ++i) { c[4*i]=cr[i].x; c[4*i+1]=cr[i].y; c[4*i+2]=cr[i].z; c[4*i+3]=cr[i].w; }
            float w[32];
#pragma unroll
            for (int m = 0; m < 16; ++m) {
                int idx = ((c[2*m] & 15) << 4) | (c[2*m+1] & 15);
                float2 p = lut[idx];
                w[2*m]   = p.x * scl;
                w[2*m+1] = p.y * scl;
            }
            __bf16* bp = &Bb[buf][brow * ROWE];
            *(v8bf*)(bp)      = cvt8(w);
            *(v8bf*)(bp + 8)  = cvt8(w + 8);
            *(v8bf*)(bp + 16) = cvt8(w + 16);
            *(v8bf*)(bp + 24) = cvt8(w + 24);

            float xv[16];
#pragma unroll
            for (int i = 0; i < 4; ++i) { xv[4*i]=xr[i].x; xv[4*i+1]=xr[i].y; xv[4*i+2]=xr[i].z; xv[4*i+3]=xr[i].w; }
            __bf16* ap = &Ab[buf][arow * ROWE + half * 16];
            *(v8bf*)(ap)     = cvt8(xv);
            *(v8bf*)(ap + 8) = cvt8(xv + 8);
        }

        // register-pipeline next tile; L2-prefetch two steps ahead
        if (kt + 1 < NKITERS) gload(kt + 1);
        if (kt + 2 < NKITERS) {
            __builtin_prefetch(crow + (kt + 2) * KSTEP, 0, 0);
            __builtin_prefetch(xrow + (kt + 2) * KSTEP, 0, 0);
        }

        __syncthreads();   // single barrier/iter is sufficient with double buffering

        // ---- WMMA compute on buf: 4x4 tiles of 16x16 per wave ----
        {
            v16bf bfrag[4];
#pragma unroll
            for (int nt = 0; nt < 4; ++nt) {
                // B operand, contiguous k order used by this code: lane n<16 holds col n, K0..15; lanes 16..31 hold K16..31
                const __bf16* bp = &Bb[buf][(wn*64 + nt*16 + (lane & 15)) * ROWE + (lane >> 4) * 16];
                v8bf lo = *(const v8bf*)(bp);
                v8bf hi = *(const v8bf*)(bp + 8);
#pragma unroll
                for (int j = 0; j < 8; ++j) { bfrag[nt][j] = lo[j]; bfrag[nt][8+j] = hi[j]; }
            }
#pragma unroll
            for (int mt = 0; mt < 4; ++mt) {
                // A operand, contiguous k order used by this code: lane<16 -> K0..15; lane>=16 -> K16..31
                const __bf16* ap = &Ab[buf][(wm*64 + mt*16 + (lane & 15)) * ROWE + (lane >> 4) * 16];
                v8bf lo = *(const v8bf*)(ap);
                v8bf hi = *(const v8bf*)(ap + 8);
                v16bf afrag;
#pragma unroll
                for (int j = 0; j < 8; ++j) { afrag[j] = lo[j]; afrag[8+j] = hi[j]; }
#pragma unroll
                for (int nt = 0; nt < 4; ++nt) {
                    acc[mt][nt] = __builtin_amdgcn_wmma_f32_16x16x32_bf16(
                        false, afrag, false, bfrag[nt], (short)0, acc[mt][nt], false, false);
                }
            }
        }
    }

    // ---- epilogue: f32 C/D layout -> row M = r + 8*(lane>=16), col N = lane&15 ----
#pragma unroll
    for (int mt = 0; mt < 4; ++mt) {
#pragma unroll
        for (int nt = 0; nt < 4; ++nt) {
#pragma unroll
            for (int r = 0; r < 8; ++r) {
                int M = t0 + wm*64 + mt*16 + r + 8*(lane >> 4);
                int N = nloc + wn*64 + nt*16 + (lane & 15);
                outp[(size_t)M * segW + N] = acc[mt][nt][r];
            }
        }
    }
}

extern "C" void kernel_launch(void* const* d_in, const int* in_sizes, int n_in,
                              void* d_out, int out_size, void* d_ws, size_t ws_size,
                              hipStream_t stream) {
    (void)n_in; (void)out_size; (void)d_ws; (void)ws_size;
    const float* x   = (const float*)d_in[0];
    const int*   qc  = (const int*)  d_in[1];
    const float* qs  = (const float*)d_in[2];
    const float* qcb = (const float*)d_in[3];
    const int*   kc  = (const int*)  d_in[4];
    const float* ks  = (const float*)d_in[5];
    const float* kcb = (const float*)d_in[6];
    const int*   vc  = (const int*)  d_in[7];
    const float* vs  = (const float*)d_in[8];
    const float* vcb = (const float*)d_in[9];
    float* out = (float*)d_out;

    const int T = in_sizes[0] / D_DIM;   // 8192 tokens
    dim3 grid((QO + KO + VO) / NTILE, T / MTILE);  // 24 x 64
    qkv_wmma_bf16_kernel<<<grid, 256, 0, stream>>>(x, qc, qs, qcb, kc, ks, kcb, vc, vs, vcb, out, T);
}
